// MiniQuixerTQ_15109694947365
// MI455X (gfx1250) — matrix operand order repaired and checked by reading, not compiled, not run on hardware
//
#include <hip/hip_runtime.h>
#include <math.h>

typedef _Float16 f16;
typedef __attribute__((ext_vector_type(16))) _Float16 v16h;
typedef __attribute__((ext_vector_type(8)))  float    v8f;

#define B_    32
#define T_    512
#define E_    1024
#define V_    32000
#define Q_    10
#define L_    8
#define DQKV_ 128
#define F_    4096   // 4*E

// ---------------------------------------------------------------------------
// 1) hmean[b,e] = mean_t (tok_emb[x[b,t],e] + pos_emb[t,e]) ; hlast = last row
// ---------------------------------------------------------------------------
__global__ __launch_bounds__(1024) void hmean_kernel(
    const int* __restrict__ x, const float* __restrict__ tok,
    const float* __restrict__ pos, float* __restrict__ hmean,
    float* __restrict__ hlast)
{
  const int b = blockIdx.x;
  const int e = threadIdx.x;
  float acc = 0.f;
  for (int t = 0; t < T_; ++t) {
    const int row = x[b * T_ + t];
    acc += tok[(size_t)row * E_ + e] + pos[t * E_ + e];
  }
  hmean[b * E_ + e] = acc * (1.f / (float)T_);
  const int last = x[b * T_ + (T_ - 1)];
  hlast[b * E_ + e] = tok[(size_t)last * E_ + e] + pos[(T_ - 1) * E_ + e];
}

// ---------------------------------------------------------------------------
// 2) context = hmean @ qkv_w[:,256:384] + b_v ; angles = pi*tanh(ctx@in_w+in_b)
// ---------------------------------------------------------------------------
__global__ __launch_bounds__(128) void context_kernel(
    const float* __restrict__ hmean, const float* __restrict__ qkv_w,
    const float* __restrict__ qkv_b, const float* __restrict__ in_w,
    const float* __restrict__ in_b, float* __restrict__ angles)
{
  __shared__ float ctx[DQKV_];
  const int b = blockIdx.x;
  const int j = threadIdx.x;
  float acc = qkv_b[2 * DQKV_ + j];
  for (int e = 0; e < E_; ++e)
    acc += hmean[b * E_ + e] * qkv_w[e * (3 * DQKV_) + 2 * DQKV_ + j];
  ctx[j] = acc;
  __syncthreads();
  if (j < Q_) {
    float a = in_b[j];
    for (int k = 0; k < DQKV_; ++k) a += ctx[k] * in_w[k * Q_ + j];
    angles[b * Q_ + j] = 3.14159265358979323846f * tanhf(a);
  }
}

// ---------------------------------------------------------------------------
// 3) PQC statevector simulation: one block per batch, 1024 complex amps in LDS
//    qubit q acts on bit k = Q-1-q. type: 0=RX, 1=RY, 2=RZ.
// ---------------------------------------------------------------------------
__global__ __launch_bounds__(512) void pqc_kernel(
    const float* __restrict__ angles, const float* __restrict__ params,
    float* __restrict__ zout)
{
  __shared__ float2 st[1 << Q_];
  __shared__ float zred[Q_];
  const int b = blockIdx.x;
  const int tid = threadIdx.x;
  st[tid]       = make_float2(tid == 0 ? 1.f : 0.f, 0.f);
  st[tid + 512] = make_float2(0.f, 0.f);
  if (tid < Q_) zred[tid] = 0.f;
  __syncthreads();

#define APPLY_GATE(kk, type, theta) do {                                        \
    const float th_ = (theta);                                                  \
    const float c_ = cosf(0.5f * th_), s_ = sinf(0.5f * th_);                   \
    const int k_ = (kk);                                                        \
    const int i0 = ((tid >> k_) << (k_ + 1)) | (tid & ((1 << k_) - 1));         \
    const int i1 = i0 | (1 << k_);                                              \
    float2 a = st[i0], d = st[i1]; float2 na, nd;                               \
    if ((type) == 0) {        /* RX: [[c,-is],[-is,c]] */                       \
      na.x =  c_*a.x + s_*d.y;  na.y =  c_*a.y - s_*d.x;                        \
      nd.x =  s_*a.y + c_*d.x;  nd.y = -s_*a.x + c_*d.y;                        \
    } else if ((type) == 1) { /* RY: [[c,-s],[s,c]] */                          \
      na.x =  c_*a.x - s_*d.x;  na.y =  c_*a.y - s_*d.y;                        \
      nd.x =  s_*a.x + c_*d.x;  nd.y =  s_*a.y + c_*d.y;                        \
    } else {                  /* RZ: diag(e^-it/2, e^+it/2) */                  \
      na.x =  c_*a.x + s_*a.y;  na.y =  c_*a.y - s_*a.x;                        \
      nd.x =  c_*d.x - s_*d.y;  nd.y =  c_*d.y + s_*d.x;                        \
    }                                                                           \
    st[i0] = na; st[i1] = nd;                                                   \
    __syncthreads();                                                            \
  } while (0)

  for (int q = 0; q < Q_; ++q) APPLY_GATE(Q_ - 1 - q, 0, angles[b * Q_ + q]);
  for (int l = 0; l < L_; ++l) {
    for (int q = 0; q < Q_; ++q) {
      const float* pp = params + ((l * Q_ + q) * 3);
      APPLY_GATE(Q_ - 1 - q, 0, pp[0]);
      APPLY_GATE(Q_ - 1 - q, 1, pp[1]);
      APPLY_GATE(Q_ - 1 - q, 2, pp[2]);
    }
    // entangling step: where bit k1 == 1, swap amplitudes across bit k2 = k1-1
    for (int q = 0; q < Q_ - 1; ++q) {
      const int k1 = Q_ - 1 - q, k2 = k1 - 1;
      if (tid < 256) {
        const int i0 = ((tid >> k2) << (k2 + 2)) | (1 << k1) |
                       (tid & ((1 << k2) - 1));
        const int i1 = i0 | (1 << k2);
        float2 t0 = st[i0]; st[i0] = st[i1]; st[i1] = t0;
      }
      __syncthreads();
    }
  }
#undef APPLY_GATE

  float zl[Q_];
#pragma unroll
  for (int q = 0; q < Q_; ++q) zl[q] = 0.f;
#pragma unroll
  for (int half = 0; half < 2; ++half) {
    const int i = tid + half * 512;
    const float2 a = st[i];
    const float pr = a.x * a.x + a.y * a.y;
#pragma unroll
    for (int q = 0; q < Q_; ++q)
      zl[q] += ((i >> (Q_ - 1 - q)) & 1) ? -pr : pr;
  }
#pragma unroll
  for (int q = 0; q < Q_; ++q) atomicAdd(&zred[q], zl[q]);
  __syncthreads();
  if (tid < Q_) zout[b * Q_ + tid] = zred[tid];
}

// ---------------------------------------------------------------------------
// 4) ln1 fused with pqc_exp = z @ map_w + map_b  (32 rows of E)
// ---------------------------------------------------------------------------
__global__ __launch_bounds__(256) void ln1_kernel(
    const float* __restrict__ hlast, const float* __restrict__ z,
    const float* __restrict__ map_w, const float* __restrict__ map_b,
    const float* __restrict__ g, const float* __restrict__ be,
    float* __restrict__ hln_f, f16* __restrict__ hln_h)
{
  const int b = blockIdx.x, tid = threadIdx.x;
  __shared__ float red[256];
  __shared__ float zz[Q_];
  if (tid < Q_) zz[tid] = z[b * Q_ + tid];
  __syncthreads();
  float xv[4]; float s = 0.f;
#pragma unroll
  for (int i = 0; i < 4; ++i) {
    const int e = tid + (i << 8);
    float pe = map_b[e];
#pragma unroll
    for (int q = 0; q < Q_; ++q) pe += zz[q] * map_w[q * E_ + e];
    const float v = hlast[b * E_ + e] + pe;
    xv[i] = v; s += v;
  }
  red[tid] = s; __syncthreads();
  for (int o = 128; o > 0; o >>= 1) { if (tid < o) red[tid] += red[tid + o]; __syncthreads(); }
  const float mean = red[0] * (1.f / E_);
  __syncthreads();
  float vs = 0.f;
#pragma unroll
  for (int i = 0; i < 4; ++i) { const float d = xv[i] - mean; vs += d * d; }
  red[tid] = vs; __syncthreads();
  for (int o = 128; o > 0; o >>= 1) { if (tid < o) red[tid] += red[tid + o]; __syncthreads(); }
  const float inv = rsqrtf(red[0] * (1.f / E_) + 1e-5f);
#pragma unroll
  for (int i = 0; i < 4; ++i) {
    const int e = tid + (i << 8);
    const float y = (xv[i] - mean) * inv * g[e] + be[e];
    hln_f[b * E_ + e] = y;
    hln_h[b * E_ + e] = (f16)y;
  }
}

// ---------------------------------------------------------------------------
// 6) ln2 over (hln + h2); emits f16 A-matrix for the vocab projection.
//    Also writes the auxiliary 0.0 scalar at the end of d_out.
// ---------------------------------------------------------------------------
__global__ __launch_bounds__(256) void ln2_kernel(
    const float* __restrict__ hln_f, const float* __restrict__ h2,
    const float* __restrict__ g, const float* __restrict__ be,
    f16* __restrict__ outh, float* __restrict__ out_tail)
{
  const int b = blockIdx.x, tid = threadIdx.x;
  __shared__ float red[256];
  float xv[4]; float s = 0.f;
#pragma unroll
  for (int i = 0; i < 4; ++i) {
    const int e = tid + (i << 8);
    const float v = hln_f[b * E_ + e] + h2[b * E_ + e];
    xv[i] = v; s += v;
  }
  red[tid] = s; __syncthreads();
  for (int o = 128; o > 0; o >>= 1) { if (tid < o) red[tid] += red[tid + o]; __syncthreads(); }
  const float mean = red[0] * (1.f / E_);
  __syncthreads();
  float vs = 0.f;
#pragma unroll
  for (int i = 0; i < 4; ++i) { const float d = xv[i] - mean; vs += d * d; }
  red[tid] = vs; __syncthreads();
  for (int o = 128; o > 0; o >>= 1) { if (tid < o) red[tid] += red[tid + o]; __syncthreads(); }
  const float inv = rsqrtf(red[0] * (1.f / E_) + 1e-5f);
#pragma unroll
  for (int i = 0; i < 4; ++i) {
    const int e = tid + (i << 8);
    outh[b * E_ + e] = (f16)((xv[i] - mean) * inv * g[e] + be[e]);
  }
  if (b == 0 && tid == 0) out_tail[0] = 0.f;  // tuple's second output: 0.0
}

// ---------------------------------------------------------------------------
// 5/7) WMMA GEMM, fully LDS-staged (A and B) with a 2-stage software pipeline.
//   Block tile: 32(M) x 256(N); 8 waves in a 2x4 grid of 16x64 wave tiles.
//   Steady-state iteration:
//     barrier; store staged regs -> LDS (s_wait_loadcnt lands HERE, one full
//     iteration after the loads were issued); barrier; issue next slab's
//     global loads; read fragments from LDS (DScnt only); 4x v_wmma.
//   So the global prefetch overlaps the whole WMMA/LDS consume phase.
//   A slab: 32x32 f16 (2 KB), one coalesced b64 per thread, conflict-free.
//   B slab: 32x256 (8192 f32 -> f16), 8 coalesced b128 per thread.
//   BNK=1 (B=(N,K) row-major, tok_emb): LDS Bs[n][k] -> b128 frag reads.
//   BNK=0 (B=(K,N) row-major, FFN weights): LDS Bs[k][n] -> u16 frag reads.
//   EPI=0: gelu(acc+bias)->f16 | EPI=1: acc+bias->f32 | EPI=2: acc->f32
//   Requires M == 32, N % 256 == 0, K % 32 == 0 (true at all call sites).
// ---------------------------------------------------------------------------
template <int BNK, int EPI>
__global__ __launch_bounds__(256) void wmma_gemm_kernel(
    const f16* __restrict__ A, const float* __restrict__ B,
    const float* __restrict__ bias, float* __restrict__ Cf,
    f16* __restrict__ Ch, int N, int K)
{
  __shared__ __align__(16) f16 Bs[256 * 32];
  __shared__ __align__(16) f16 As[32 * 32];
  const int tid   = threadIdx.x;
  const int wave  = tid >> 5;
  const int lane  = tid & 31;
  const int group = lane >> 4;   // half-wave
  const int lr    = lane & 15;
  const int mi    = wave & 1;    // 2 M-tiles
  const int ni    = wave >> 1;   // 4 N-subtiles of 64
  const int nbase = blockIdx.x << 8;
  const int mbase = mi << 4;

  // A-slab cooperative mapping: thread -> (row mA, k-quad kkA)
  const int mA  = tid >> 3;        // 0..31
  const int kkA = (tid & 7) << 2;  // 0,4,...,28

  v8f acc[4] = {};

  float4 stg[8];
  uint2  astg;
  auto load_stage = [&](int kb) {
    astg = *(const uint2*)(A + (size_t)mA * K + kb + kkA);  // 4 f16, coalesced
#pragma unroll
    for (int i = 0; i < 8; ++i) {
      const int linear = (i * 256 + tid) * 4;   // 8192 f32 per slab
      if (BNK == 1) {
        const int nn = linear >> 5;             // 0..255
        const int kk = linear & 31;             // 0..28 step 4
        stg[i] = *(const float4*)(B + (size_t)(nbase + nn) * K + kb + kk);
      } else {
        const int kk = linear >> 8;             // 0..31
        const int nn = linear & 255;            // 0..252 step 4
        stg[i] = *(const float4*)(B + (size_t)(kb + kk) * N + nbase + nn);
      }
    }
  };
  auto store_stage = [&]() {
    *(uint2*)&As[mA * 32 + kkA] = astg;         // conflict-free b64
#pragma unroll
    for (int i = 0; i < 8; ++i) {
      const int linear = (i * 256 + tid) * 4;
      f16* p;
      if (BNK == 1) {
        const int nn = linear >> 5;
        const int kk = linear & 31;
        p = &Bs[nn * 32 + kk];                  // [n][k]
      } else {
        const int kk = linear >> 8;
        const int nn = linear & 255;
        p = &Bs[kk * 256 + nn];                 // [k][n]
      }
      p[0] = (f16)stg[i].x; p[1] = (f16)stg[i].y;
      p[2] = (f16)stg[i].z; p[3] = (f16)stg[i].w;
    }
  };

  load_stage(0);
  for (int kb = 0; kb < K; kb += 32) {
    __syncthreads();            // previous slab's LDS reads complete
    store_stage();              // consumes staged regs (waits on loads here)
    __syncthreads();            // slab visible to all waves
    if (kb + 32 < K) load_stage(kb + 32);  // prefetch overlaps compute below

    // A fragment: per lane one contiguous 32B chunk, k = 16*group + s (matches B)
    v16h a;
#pragma unroll
    for (int v = 0; v < 8; ++v) {
      const int koff = (group << 4) + (v << 1);
      const f16* ap = &As[(mbase + lr) * 32 + koff];
      a[2 * v]     = ap[0];
      a[2 * v + 1] = ap[1];
    }
#pragma unroll
    for (int j = 0; j < 4; ++j) {
      const int nloc = (ni << 6) + j * 16 + lr;
      v16h bf;
#pragma unroll
      for (int e = 0; e < 16; ++e) {
        bf[e] = (BNK == 1) ? Bs[nloc * 32 + (group << 4) + e]
                           : Bs[((group << 4) + e) * 256 + nloc];
      }
      acc[j] = __builtin_amdgcn_wmma_f32_16x16x32_f16(
          /*neg_a=*/false, a, /*neg_b=*/false, bf,
          /*c_mod=*/(short)0, acc[j], /*reuse_a=*/false, /*reuse_b=*/false);
    }
  }

  // D layout: VGPR r -> M = mbase + group*8 + r ; N = nbase + ni*64 + j*16 + lr
#pragma unroll
  for (int j = 0; j < 4; ++j) {
    const int n = nbase + (ni << 6) + j * 16 + lr;
    const float bv = (EPI == 2) ? 0.f : bias[n];
#pragma unroll
    for (int r = 0; r < 8; ++r) {
      const int m = mbase + (group << 3) + r;
      float xv = acc[j][r] + bv;
      if (EPI == 0) {
        xv = 0.5f * xv * (1.f + erff(xv * 0.70710678118654752f)); // exact GELU
        Ch[(size_t)m * N + n] = (f16)xv;
      } else {
        Cf[(size_t)m * N + n] = xv;
      }
    }
  }
}

// ---------------------------------------------------------------------------
extern "C" void kernel_launch(void* const* d_in, const int* in_sizes, int n_in,
                              void* d_out, int out_size, void* d_ws, size_t ws_size,
                              hipStream_t stream) {
  const int*   x     = (const int*)d_in[0];
  const float* tok   = (const float*)d_in[1];
  const float* pos   = (const float*)d_in[2];
  const float* qkv_w = (const float*)d_in[3];
  const float* qkv_b = (const float*)d_in[4];
  const float* in_w  = (const float*)d_in[5];
  const float* in_b  = (const float*)d_in[6];
  const float* pqc_p = (const float*)d_in[7];
  const float* map_w = (const float*)d_in[8];
  const float* map_b = (const float*)d_in[9];
  const float* w1    = (const float*)d_in[10];
  const float* b1    = (const float*)d_in[11];
  const float* w2    = (const float*)d_in[12];
  const float* b2    = (const float*)d_in[13];
  const float* ln1g  = (const float*)d_in[14];
  const float* ln1b  = (const float*)d_in[15];
  const float* ln2g  = (const float*)d_in[16];
  const float* ln2b  = (const float*)d_in[17];
  float* out = (float*)d_out;

  char* w = (char*)d_ws;
  auto alloc = [&](size_t bytes) -> void* {
    void* p = (void*)w;
    w += (bytes + 255) & ~(size_t)255;
    return p;
  };
  float* hmean  = (float*)alloc((size_t)B_ * E_ * 4);
  float* hlast  = (float*)alloc((size_t)B_ * E_ * 4);
  float* angles = (float*)alloc((size_t)B_ * Q_ * 4);
  float* zbuf   = (float*)alloc((size_t)B_ * Q_ * 4);
  float* hln_f  = (float*)alloc((size_t)B_ * E_ * 4);
  f16*   hln_h  = (f16*)  alloc((size_t)B_ * E_ * 2);
  f16*   act_h  = (f16*)  alloc((size_t)B_ * F_ * 2);
  float* h2_f   = (float*)alloc((size_t)B_ * E_ * 4);
  f16*   h2ln_h = (f16*)  alloc((size_t)B_ * E_ * 2);

  // 1) embedding mean over T + last-token row
  hmean_kernel<<<B_, 1024, 0, stream>>>(x, tok, pos, hmean, hlast);
  // 2) context -> angles
  context_kernel<<<B_, 128, 0, stream>>>(hmean, qkv_w, qkv_b, in_w, in_b, angles);
  // 3) quantum circuit simulation -> z
  pqc_kernel<<<B_, 512, 0, stream>>>(angles, pqc_p, zbuf);
  // 4) ln1(h_last + z@map_w + map_b)
  ln1_kernel<<<B_, 256, 0, stream>>>(hlast, zbuf, map_w, map_b, ln1g, ln1b,
                                     hln_f, hln_h);
  // 5) FFN1: gelu(hln @ W1 + b1) -> f16 act       (M=32, N=4096, K=1024)
  wmma_gemm_kernel<0, 0><<<F_ / 256, 256, 0, stream>>>(
      hln_h, w1, b1, nullptr, act_h, F_, E_);
  // 5b) FFN2: act @ W2 + b2 -> f32 h2             (M=32, N=1024, K=4096)
  wmma_gemm_kernel<0, 1><<<E_ / 256, 256, 0, stream>>>(
      act_h, w2, b2, h2_f, nullptr, E_, F_);
  // 6) ln2(hln + h2) -> f16 ; also writes out[B*V] = 0.0
  ln2_kernel<<<B_, 256, 0, stream>>>(hln_f, h2_f, ln2g, ln2b, h2ln_h,
                                     out + (size_t)B_ * V_);
  // 7) vocab projection: h2ln @ tok_emb^T -> out  (M=32, N=32000, K=1024)
  wmma_gemm_kernel<1, 2><<<V_ / 256, 256, 0, stream>>>(
      h2ln_h, tok, nullptr, out, nullptr, V_, E_);
}
